// BaseModel_72275709657397
// MI455X (gfx1250) — matrix operand order repaired and checked by reading, not compiled, not run on hardware
//
#include <hip/hip_runtime.h>

// ---------------------------------------------------------------------------
// VQ-VAE nearest-codebook lookup for MI455X (gfx1250, wave32, WMMA).
//
// Score GEMM ze[16384x512] x cb^T[512x4096] = 68.7 GFLOP vs ~75MB traffic ->
// compute-bound by ~100x, so push it through v_wmma_f32_16x16x32_bf16.
//  * codebook pre-converted to bf16 ONCE (prep kernel, d_ws); B-tile staging
//    via CDNA5 async global->LDS path (global_load_async_to_lds_b128,
//    ASYNCcnt) - emission confirmed in round 3.
//  * A fragments register-resident (16 x v16bf = 128 VGPRs) across the whole
//    4096-entry loop -> 2 ds_load_b128 per WMMA instead of 4. Round-3 asm
//    showed the allocator spilling these to scratch under its default
//    occupancy heuristic -> __launch_bounds__(256, 2) raises the per-wave
//    VGPR budget (kernel is LDS-capped to ~3 blocks/WGP anyway).
//  * running argmin kept in registers as (orderkey<<32)|entry u64; shuffle
//    reduction + LDS ds_min_u64 happens once at the end.
// Workspace: norms 16KB @0, bf16 codebook 4MB @16KB.
// ---------------------------------------------------------------------------

typedef __attribute__((ext_vector_type(16))) __bf16 v16bf;
typedef __attribute__((ext_vector_type(8)))  float  v8f;
typedef __attribute__((ext_vector_type(4)))  int    v4i;

#define C_DIM    512
#define K_CB     4096
#define M_TILE   32
#define N_TILE   64
#define A_STRIDE 520   // bf16 elems per A row (pad -> 1040B, 16B aligned)
#define B_STRIDE 520   // bf16 elems per B row
#define Z_STRIDE 516   // f32 elems per epilogue row (2064B, 16B aligned)

#if defined(__has_builtin)
#if __has_builtin(__builtin_amdgcn_global_load_async_to_lds_b128) && \
    __has_builtin(__builtin_amdgcn_s_wait_asynccnt)
#define USE_ASYNC_LDS 1
#endif
#endif
#ifndef USE_ASYNC_LDS
#define USE_ASYNC_LDS 0
#endif

__device__ __forceinline__ unsigned short f32_to_bf16(float f) {
  unsigned int b = __float_as_uint(f);
  unsigned int r = b + 0x7FFFu + ((b >> 16) & 1u);   // round-to-nearest-even
  return (unsigned short)(r >> 16);
}

// ---------------- prep: codebook f32 -> bf16 + row norms -------------------
__global__ void cb_prep_kernel(const float* __restrict__ cb,
                               unsigned short* __restrict__ cbh,
                               float* __restrict__ norms) {
  const int row = blockIdx.x;            // 0..4095
  const int t   = threadIdx.x;           // 256 threads
  const float2 v = ((const float2*)(cb + (size_t)row * C_DIM))[t];
  unsigned int packed = (unsigned int)f32_to_bf16(v.x) |
                        ((unsigned int)f32_to_bf16(v.y) << 16);
  ((unsigned int*)(cbh + (size_t)row * C_DIM))[t] = packed;
  float s = v.x * v.x + v.y * v.y;
  for (int off = 16; off > 0; off >>= 1) s += __shfl_xor(s, off, 32);
  __shared__ float partial[8];
  if ((t & 31) == 0) partial[t >> 5] = s;
  __syncthreads();
  if (t == 0) {
    float tot = 0.f;
#pragma unroll
    for (int i = 0; i < 8; ++i) tot += partial[i];
    norms[row] = tot;
  }
}

// ------------------------------ main kernel --------------------------------
__global__ void __launch_bounds__(256, 2)
vq_argmin_kernel(const float* __restrict__ feature,
                 const float* __restrict__ cb,           // f32 (epilogue gather)
                 const unsigned short* __restrict__ cbh, // bf16
                 const float* __restrict__ norms,
                 float* __restrict__ out) {
  extern __shared__ unsigned char smem[];
  unsigned short* A  = (unsigned short*)smem;             // [32][520] bf16
  unsigned short* B  = (unsigned short*)(smem + 33280);   // [64][520] bf16
  float*          ZQ = (float*)smem;                      // [16][516] (epilogue)
  float*          ZE = (float*)(smem + 33024);            // [16][516] (epilogue)
  __shared__ unsigned long long best[M_TILE];

  const int tid  = threadIdx.x;
  const int lane = tid & 31;
  const int wv   = tid >> 5;       // wave 0..7
  const int wm   = wv >> 2;        // 0..1  (M sub-tile)
  const int wn   = wv & 3;         // 0..3  (N sub-tile)
  const int half = lane >> 4;
  const int l16  = lane & 15;

  const int t0    = blockIdx.x * M_TILE;   // first token of this block
  const int n_img = t0 >> 12;              // 4096 tokens per image
  const int hw0   = t0 & 4095;
  const float* fbase = feature + (size_t)n_img * C_DIM * 4096 + hw0;

  if (tid < M_TILE) best[tid] = ~0ull;

  // ---- stage A tile: 32 tokens x 512 ch, f32 -> bf16 (coalesced over m) ----
#pragma unroll 4
  for (int i = 0; i < 64; ++i) {
    int flat = tid + i * 256;              // 0..16383
    int c = flat >> 5;
    int m = flat & 31;
    A[m * A_STRIDE + c] = f32_to_bf16(fbase[(size_t)c * 4096 + m]);
  }
  __syncthreads();

  // ---- A fragments register-resident for the whole K loop (128 VGPRs) ----
  const unsigned short* Arow = A + (wm * 16 + l16) * A_STRIDE;
  v16bf afv[16];
#pragma unroll
  for (int kc = 0; kc < 16; ++kc) {
    union { v16bf v; uint4 u[2]; } t;
    // A (16x32): lane = row M; VGPR0-3 hold K = 8*half..+7, VGPR4-7 K+16
    t.u[0] = *(const uint4*)(Arow + kc * 32 + 8 * half);
    t.u[1] = *(const uint4*)(Arow + kc * 32 + 16 + 8 * half);
    afv[kc] = t.v;
  }

  const unsigned short* Brow = B + (wn * 16 + l16) * B_STRIDE;
  unsigned long long bestkk[8];
#pragma unroll
  for (int r = 0; r < 8; ++r) bestkk[r] = ~0ull;

  for (int nb = 0; nb < K_CB / N_TILE; ++nb) {
    const int k0 = nb * N_TILE;

    // ---- stage B tile: 64 entries x 512 ch of pre-converted bf16 ----
    const unsigned short* srcb = cbh + (size_t)k0 * C_DIM;
#pragma unroll
    for (int i = 0; i < 16; ++i) {
      int flat = tid + i * 256;            // 16B unit index, 0..4095
      int row  = flat >> 6;                // 64 units per row
      int c8   = flat & 63;
      const uint4* gp = (const uint4*)(srcb + (size_t)row * C_DIM + c8 * 8);
      uint4*       lp = (uint4*)(B + row * B_STRIDE + c8 * 8);
#if USE_ASYNC_LDS
      __builtin_amdgcn_global_load_async_to_lds_b128(
          (__attribute__((address_space(1))) v4i*)(uintptr_t)gp,
          (__attribute__((address_space(3))) v4i*)(unsigned)(uintptr_t)lp,
          0, 0);
#else
      *lp = *gp;
#endif
    }
#if USE_ASYNC_LDS
    __builtin_amdgcn_s_wait_asynccnt(0);
#endif
    __syncthreads();

    const float cn = norms[k0 + wn * 16 + l16];   // my 16 columns' ||cb||^2

    // ---- 16x16x32 bf16 WMMA over full C dim (16 k-steps) ----
    v8f acc = {};
#pragma unroll
    for (int kc = 0; kc < 16; ++kc) {
      const int c0 = kc * 32;
      union { v16bf v; uint4 u[2]; } bfm;
      // B (32x16): lane%16 = col N; VGPR0-3 hold K = 8*half..+7, VGPR4-7 K+16
      bfm.u[0] = *(const uint4*)(Brow + c0 + 8 * half);
      bfm.u[1] = *(const uint4*)(Brow + c0 + 16 + 8 * half);
      acc = __builtin_amdgcn_wmma_f32_16x16x32_bf16(
          false, afv[kc], false, bfm.v, (short)0, acc, false, false);
    }

    // ---- running argmin in registers (2 ops per row per iteration) ----
    // C/D layout: VGPR r, lanes 0-15 -> row r, lanes 16-31 -> row r+8; N=l16
    const unsigned int entry = (unsigned int)(k0 + wn * 16 + l16);
#pragma unroll
    for (int r = 0; r < 8; ++r) {
      float d = cn - 2.0f * acc[r];
      unsigned int bits = __float_as_uint(d);
      unsigned int key  = ((int)bits < 0) ? ~bits : (bits | 0x80000000u);
      unsigned long long kk = ((unsigned long long)key << 32) | entry;
      if (kk < bestkk[r]) bestkk[r] = kk;
    }
    __syncthreads();   // B tile rewritten next iteration
  }

  // ---- final cross-lane / cross-wave argmin reduction (once) ----
#pragma unroll
  for (int r = 0; r < 8; ++r) {
    unsigned long long kk = bestkk[r];
#pragma unroll
    for (int off = 1; off < 16; off <<= 1) {
      unsigned int ohi = __shfl_xor((unsigned int)(kk >> 32), off, 32);
      unsigned int olo = __shfl_xor((unsigned int)kk, off, 32);
      unsigned long long o = ((unsigned long long)ohi << 32) | olo;
      if (o < kk) kk = o;
    }
    if (l16 == 0)
      atomicMin(&best[wm * 16 + r + 8 * half], kk);   // ds_min_u64
  }
  __syncthreads();

  // ------------------- epilogue: out = ze + (zq - ze) ----------------------
  for (int hblk = 0; hblk < 2; ++hblk) {
    const int mbase = hblk * 16;
    {
      // gather winning codebook rows in f32 (coalesced 256B per 16-lane group)
      int m  = tid >> 4;                       // 0..15
      int jl = tid & 15;
      int bi = (int)(unsigned int)best[mbase + m];   // low 32b = entry idx
      const float4* src = (const float4*)(cb + (size_t)bi * C_DIM);
#pragma unroll
      for (int it = 0; it < 8; ++it) {
        int c4 = jl + it * 16;
        *(float4*)(ZQ + m * Z_STRIDE + c4 * 4) = src[c4];
      }
    }
    // re-read ze in f32, coalesced over m, staged for transposed writeback
#pragma unroll 4
    for (int i = 0; i < 32; ++i) {
      int flat = tid + i * 256;                // 16*512 elems
      int c = flat >> 4;
      int m = flat & 15;
      ZE[m * Z_STRIDE + c] = fbase[(size_t)c * 4096 + mbase + m];
    }
    __syncthreads();
    {
      // coalesced 128B stores of ze + (zq - ze)  (straight-through estimator)
      int m  = tid >> 4;
      int jl = tid & 15;
      float* orow = out + (size_t)(t0 + mbase + m) * C_DIM;
#pragma unroll
      for (int it = 0; it < 8; ++it) {
        int c = (jl + it * 16) * 4;
        float4 o;
        float ze0 = ZE[m * Z_STRIDE + c + 0], zq0 = ZQ[m * Z_STRIDE + c + 0];
        float ze1 = ZE[m * Z_STRIDE + c + 1], zq1 = ZQ[m * Z_STRIDE + c + 1];
        float ze2 = ZE[m * Z_STRIDE + c + 2], zq2 = ZQ[m * Z_STRIDE + c + 2];
        float ze3 = ZE[m * Z_STRIDE + c + 3], zq3 = ZQ[m * Z_STRIDE + c + 3];
        o.x = ze0 + (zq0 - ze0);
        o.y = ze1 + (zq1 - ze1);
        o.z = ze2 + (zq2 - ze2);
        o.w = ze3 + (zq3 - ze3);
        *(float4*)(orow + c) = o;
      }
    }
    __syncthreads();
  }
}

// ------------------------------- launcher ----------------------------------
extern "C" void kernel_launch(void* const* d_in, const int* in_sizes, int n_in,
                              void* d_out, int out_size, void* d_ws, size_t ws_size,
                              hipStream_t stream) {
  const float* feature = (const float*)d_in[0];   // [4, 512, 64, 64] f32
  const float* cbw     = (const float*)d_in[1];   // [4096, 512] f32
  float* out = (float*)d_out;                     // [16384, 512] f32

  // workspace: [0,16KB) row norms f32; [16KB, 16KB+4MB) bf16 codebook
  float*          norms = (float*)d_ws;
  unsigned short* cbh   = (unsigned short*)((char*)d_ws + 16384);

  cb_prep_kernel<<<K_CB, 256, 0, stream>>>(cbw, cbh, norms);

  // dynamic LDS: A(33280) + B(64*520*2=66560) = 99840 bytes
  const size_t shmem = 99840;
  vq_argmin_kernel<<<16384 / M_TILE, 256, shmem, stream>>>(feature, cbw, cbh,
                                                           norms, out);
}
